// GraphSAGEModel_57131654971944
// MI455X (gfx1250) — matrix operand order repaired and checked by reading, not compiled, not run on hardware
//
#include <hip/hip_runtime.h>

#define NN 100000
#define NE 640000
#define D  128
#define K2 256

typedef __attribute__((ext_vector_type(16))) __bf16 v16bf;
typedef __attribute__((ext_vector_type(8)))  __bf16 v8bf;
typedef __attribute__((ext_vector_type(8)))  float  v8f;

// ---------------- utility: zero an f32 buffer ----------------
__global__ __launch_bounds__(256) void zero_f32(float* __restrict__ p, int n) {
  int i = blockIdx.x * 256 + threadIdx.x;
  if (i < n) p[i] = 0.0f;
}

// ---------------- degree (in-degree count, f32) ----------------
__global__ __launch_bounds__(256) void degree_kernel(const int* __restrict__ dst,
                                                     float* __restrict__ cnt) {
  int e = blockIdx.x * 256 + threadIdx.x;
  if (e < NE) atomicAdd(&cnt[dst[e]], 1.0f);
}

// ------------- gather x[src] and scatter-add onto sum[dst] -------------
// one thread per (edge, 4-feature chunk); float4 gather, 4 f32 atomics.
// sum[] working set (51 MB) is L2-resident on MI455X (192 MB L2), so the
// atomics resolve in L2.
__global__ __launch_bounds__(256) void scatter_kernel(const float* __restrict__ feat,
                                                      const int* __restrict__ src,
                                                      const int* __restrict__ dst,
                                                      float* __restrict__ sum) {
  int gid = blockIdx.x * 256 + threadIdx.x;
  int e = gid >> 5;
  if (e >= NE) return;
  int q = gid & 31;
  int s = src[e], d = dst[e];
  float4 v = ((const float4*)(feat + (size_t)s * D))[q];
  float* o = sum + (size_t)d * D + (q << 2);
  atomicAdd(o + 0, v.x);
  atomicAdd(o + 1, v.y);
  atomicAdd(o + 2, v.z);
  atomicAdd(o + 3, v.w);
}

// ------------- build bf16 A = [mean | self] (N x 256) -------------
__global__ __launch_bounds__(256) void pack_a_kernel(const float* __restrict__ sum,
                                                     const float* __restrict__ cnt,
                                                     const float* __restrict__ self,
                                                     __bf16* __restrict__ Abf) {
  int gid = blockIdx.x * 256 + threadIdx.x;
  int n = gid >> 5;
  if (n >= NN) return;
  int q = gid & 31;
  float r = 1.0f / fmaxf(cnt[n], 1.0f);
  float4 s = ((const float4*)(sum  + (size_t)n * D))[q];
  float4 x = ((const float4*)(self + (size_t)n * D))[q];
  union { __bf16 b[4]; uint2 u; } pm, px;
  pm.b[0] = (__bf16)(s.x * r); pm.b[1] = (__bf16)(s.y * r);
  pm.b[2] = (__bf16)(s.z * r); pm.b[3] = (__bf16)(s.w * r);
  px.b[0] = (__bf16)x.x; px.b[1] = (__bf16)x.y;
  px.b[2] = (__bf16)x.z; px.b[3] = (__bf16)x.w;
  *(uint2*)(Abf + (size_t)n * K2 + (q << 2))     = pm.u;
  *(uint2*)(Abf + (size_t)n * K2 + D + (q << 2)) = px.u;
}

// ------------- pre-permute W = [W_l ; W_r] (256x128) into WMMA B fragments ----
// frag = kt*8 + nt (kt: K/32 tile, nt: N/16 tile); within a fragment each
// lane's 16 bf16 K-values are contiguous so the GEMM does one 32B load/lane.
// Packed B layout (wave32, 16-bit, 32x16; contiguous K, not ISA table order): lane holds column n = lane&15,
// K base = (lane<16 ? 0 : 16), VGPR j packs B[kb+2j] (lo) / B[kb+2j+1] (hi).
__global__ __launch_bounds__(256) void pack_w_kernel(const float* __restrict__ Wl,
                                                     const float* __restrict__ Wr,
                                                     __bf16* __restrict__ Wp) {
  int gid = blockIdx.x * 256 + threadIdx.x;
  if (gid >= 64 * 32) return;
  int lane = gid & 31;
  int frag = gid >> 5;
  int kt = frag >> 3;
  int nt = frag & 7;
  int n  = nt * 16 + (lane & 15);
  int kb = kt * 32 + ((lane < 16) ? 0 : 16);
  __bf16* out = Wp + (size_t)frag * 512 + lane * 16;
#pragma unroll
  for (int j = 0; j < 16; ++j) {
    int k = kb + j;
    float w = (k < D) ? Wl[(size_t)k * D + n] : Wr[(size_t)(k - D) * D + n];
    out[j] = (__bf16)w;
  }
}

// ------------- fused GEMM: C = relu(A @ [W_l;W_r] + b_l) --------------
// block = 256 threads = 8 waves; block owns 16 rows of A (staged in LDS,
// reused by all 8 waves), wave w owns output columns [16w, 16w+16).
// K = 256 -> 8 x v_wmma_f32_16x16x32_bf16 per wave.
__global__ __launch_bounds__(256) void sage_gemm_kernel(const __bf16* __restrict__ A,
                                                        const __bf16* __restrict__ Bp,
                                                        const float*  __restrict__ bias,
                                                        float* __restrict__ C) {
  __shared__ __bf16 As[16 * 264];   // 256 cols + 8-elem pad: bank-offset rows
  const int tid = threadIdx.x;
  const int mt  = blockIdx.x;
  {
    // 16 rows x 256 bf16 = 8 KB contiguous in A; thread = (row, 32B chunk)
    int row = tid >> 4, chunk = tid & 15;
    const uint4* g = (const uint4*)(A + (size_t)mt * (16 * K2) + row * K2 + chunk * 16);
    uint4* s = (uint4*)(As + row * 264 + chunk * 16);
    s[0] = g[0];
    s[1] = g[1];
  }
  __syncthreads();
  const int wave = tid >> 5;             // n-tile 0..7
  const int lane = tid & 31;
  const int mrow = lane & 15;
  const int kb   = (lane < 16) ? 0 : 16; // contiguous A-fragment K base (matches packed B)
  v8f acc = {};
#pragma unroll
  for (int kt = 0; kt < 8; ++kt) {
    union { v16bf v; v8bf h[2]; } a;
    const __bf16* ap = As + mrow * 264 + kt * 32 + kb;
    a.h[0] = *(const v8bf*)ap;           // K = kb .. kb+7
    a.h[1] = *(const v8bf*)(ap + 8);     // K = kb+8 .. kb+15
    v16bf b = *(const v16bf*)(Bp + (size_t)(kt * 8 + wave) * 512 + lane * 16);
    acc = __builtin_amdgcn_wmma_f32_16x16x32_bf16(false, a.v, false, b,
                                                  (short)0, acc, false, false);
  }
  // epilogue: bias + ReLU.  C/D layout: VGPR r -> M = r + (lane<16 ? 0 : 8)
  const int ncol = (wave << 4) + (lane & 15);
  const float bv = bias[ncol];
  const int m0 = (mt << 4) + ((lane < 16) ? 0 : 8);
#pragma unroll
  for (int r = 0; r < 8; ++r) {
    float v = acc[r] + bv;
    C[(size_t)(m0 + r) * D + ncol] = v > 0.0f ? v : 0.0f;
  }
}

extern "C" void kernel_launch(void* const* d_in, const int* in_sizes, int n_in,
                              void* d_out, int out_size, void* d_ws, size_t ws_size,
                              hipStream_t stream) {
  const float* x   = (const float*)d_in[0];
  const int*   ei  = (const int*)d_in[1];
  const float* Wl0 = (const float*)d_in[2];
  const float* bl0 = (const float*)d_in[3];
  const float* Wr0 = (const float*)d_in[4];
  const float* Wl1 = (const float*)d_in[5];
  const float* bl1 = (const float*)d_in[6];
  const float* Wr1 = (const float*)d_in[7];
  const int* src = ei;            // edge_index[0]
  const int* dst = ei + NE;       // edge_index[1]

  char* ws = (char*)d_ws;
  size_t off = 0;
  float*  sum = (float*)(ws + off);  off += (size_t)NN * D * 4;   // 51.2 MB
  float*  cnt = (float*)(ws + off);  off += 512 * 1024;           // 0.5 MB
  float*  h   = (float*)(ws + off);  off += (size_t)NN * D * 4;   // 51.2 MB
  __bf16* Abf = (__bf16*)(ws + off); off += (size_t)NN * K2 * 2;  // 51.2 MB
  __bf16* Wp0 = (__bf16*)(ws + off); off += 64 * 512 * 2;         // 64 KB
  __bf16* Wp1 = (__bf16*)(ws + off); off += 64 * 512 * 2;         // 64 KB

  const int ZB_SUM = (NN * D + 255) / 256;   // 50000
  const int ZB_CNT = (NN + 255) / 256;       // 391
  const int EB     = (NE + 255) / 256;       // 2500
  const int SB     = (NE * 32) / 256;        // 80000
  const int PB     = (NN * 32) / 256;        // 12500
  const int GB     = NN / 16;                // 6250 (exact)

  // ---- layer 0 ----
  zero_f32<<<ZB_SUM, 256, 0, stream>>>(sum, NN * D);
  zero_f32<<<ZB_CNT, 256, 0, stream>>>(cnt, NN);
  degree_kernel<<<EB, 256, 0, stream>>>(dst, cnt);          // same for both layers
  scatter_kernel<<<SB, 256, 0, stream>>>(x, src, dst, sum);
  pack_w_kernel<<<8, 256, 0, stream>>>(Wl0, Wr0, Wp0);
  pack_w_kernel<<<8, 256, 0, stream>>>(Wl1, Wr1, Wp1);
  pack_a_kernel<<<PB, 256, 0, stream>>>(sum, cnt, x, Abf);
  sage_gemm_kernel<<<GB, 256, 0, stream>>>(Abf, Wp0, bl0, h);

  // ---- layer 1 ----
  zero_f32<<<ZB_SUM, 256, 0, stream>>>(sum, NN * D);
  scatter_kernel<<<SB, 256, 0, stream>>>(h, src, dst, sum);
  pack_a_kernel<<<PB, 256, 0, stream>>>(sum, cnt, h, Abf);
  sage_gemm_kernel<<<GB, 256, 0, stream>>>(Abf, Wp1, bl1, (float*)d_out);

  (void)in_sizes; (void)n_in; (void)out_size; (void)ws_size;
}
